// BigramLanguageModel_73581379715740
// MI455X (gfx1250) — compile-verified, matrix operand order checked by reading, not run on hardware
//
#include <hip/hip_runtime.h>

// ---------------- model constants ----------------
#define Bc   4
#define Tc   1024
#define Ec   1024
#define Hc   16
#define Dc   64
#define Lc   8
#define FFc  4096
#define Vc   50257
#define BTc  (Bc * Tc)
#define EPSc 1e-5f

typedef __bf16 v16bf  __attribute__((ext_vector_type(16)));
typedef __bf16 bf16x8 __attribute__((ext_vector_type(8)));
typedef float  v8f    __attribute__((ext_vector_type(8)));
typedef int    v4i    __attribute__((ext_vector_type(4)));

#if defined(__has_builtin)
#if __has_builtin(__builtin_amdgcn_global_load_async_to_lds_b128)
#define HAS_ASYNC_LDS 1
#endif
#endif

#ifdef HAS_ASYNC_LDS
typedef __attribute__((address_space(1))) v4i as1_v4i;
typedef __attribute__((address_space(3))) v4i as3_v4i;
__device__ __forceinline__ void async_b128(const __bf16* src, __bf16* lds_dst) {
  __builtin_amdgcn_global_load_async_to_lds_b128(
      (as1_v4i*)(__bf16*)src, (as3_v4i*)lds_dst, 0, 0);
}
__device__ __forceinline__ void wait_async2() {
#if __has_builtin(__builtin_amdgcn_s_wait_asynccnt)
  __builtin_amdgcn_s_wait_asynccnt(2);
#else
  asm volatile("s_wait_asynccnt 0x2" ::: "memory");
#endif
}
__device__ __forceinline__ void wait_async0() {
#if __has_builtin(__builtin_amdgcn_s_wait_asynccnt)
  __builtin_amdgcn_s_wait_asynccnt(0);
#else
  asm volatile("s_wait_asynccnt 0x0" ::: "memory");
#endif
}
#endif

// CDNA5 16-bit A/B fragment: lanes 0-15 carry K {0..7,16..23}, lanes 16-31
// carry K {8..15,24..31}. Each half is a contiguous 16-byte group -> two
// b128 loads per fragment.
__device__ __forceinline__ v16bf frag_ld(const __bf16* p, bool hi) {
  int base = hi ? 8 : 0;
  bf16x8 lo = *(const bf16x8*)(p + base);
  bf16x8 hh = *(const bf16x8*)(p + base + 16);
  return __builtin_shufflevector(lo, hh, 0, 1, 2, 3, 4, 5, 6, 7,
                                 8, 9, 10, 11, 12, 13, 14, 15);
}

__device__ __forceinline__ v8f wmma_bf16(v16bf a, v16bf b, v8f c) {
  return __builtin_amdgcn_wmma_f32_16x16x32_bf16(
      false, a, false, b, (short)0, c, false, false);
}

// ---------------- embedding ----------------
__global__ __launch_bounds__(256) void embed_kernel(
    const int* __restrict__ idx, const float* __restrict__ tok,
    const float* __restrict__ pos, float* __restrict__ x) {
  int bt = blockIdx.x;
  int t  = bt & (Tc - 1);
  int tokid = idx[bt];
  const float* te = tok + (size_t)tokid * Ec;
  const float* pe = pos + (size_t)t * Ec;
  float* xo = x + (size_t)bt * Ec;
  for (int i = threadIdx.x; i < Ec; i += 256) xo[i] = te[i] + pe[i];
}

// ---------------- layernorm: fp32 in -> bf16 out ----------------
__global__ __launch_bounds__(256) void ln_kernel(
    const float* __restrict__ x, const float* __restrict__ g,
    const float* __restrict__ b, __bf16* __restrict__ out) {
  __shared__ float red[256];
  int row = blockIdx.x, tid = threadIdx.x;
  const float* xr = x + (size_t)row * Ec;

  float s = 0.f;
  for (int i = tid; i < Ec; i += 256) s += xr[i];
  red[tid] = s; __syncthreads();
  for (int o = 128; o > 0; o >>= 1) { if (tid < o) red[tid] += red[tid + o]; __syncthreads(); }
  float mu = red[0] * (1.0f / Ec);
  __syncthreads();

  float v = 0.f;
  for (int i = tid; i < Ec; i += 256) { float d = xr[i] - mu; v += d * d; }
  red[tid] = v; __syncthreads();
  for (int o = 128; o > 0; o >>= 1) { if (tid < o) red[tid] += red[tid + o]; __syncthreads(); }
  float rstd = rsqrtf(red[0] * (1.0f / Ec) + EPSc);

  __bf16* orow = out + (size_t)row * Ec;
  for (int i = tid; i < Ec; i += 256)
    orow[i] = (__bf16)((xr[i] - mu) * rstd * g[i] + b[i]);
}

// ---------------- weight transpose + fp32->bf16: in[K,N] -> out[N,K] ----------------
// grid (ceil(N/32), K/32, batch); block (32,8)
__global__ __launch_bounds__(256) void transpose_cvt(
    const float* __restrict__ in, __bf16* __restrict__ out,
    int K, int N, size_t in_bs, size_t out_bs) {
  __shared__ float t[32][33];
  in  += (size_t)blockIdx.z * in_bs;
  out += (size_t)blockIdx.z * out_bs;
  int nt = blockIdx.x * 32, kt = blockIdx.y * 32;
  int tx = threadIdx.x, ty = threadIdx.y;
#pragma unroll
  for (int i = 0; i < 4; i++) {
    int k = kt + ty + 8 * i;
    int n = nt + tx;
    t[ty + 8 * i][tx] = (n < N) ? in[(size_t)k * N + n] : 0.f;
  }
  __syncthreads();
#pragma unroll
  for (int i = 0; i < 4; i++) {
    int n = nt + ty + 8 * i;
    int k = kt + tx;
    if (n < N) out[(size_t)n * K + k] = (__bf16)t[tx][ty + 8 * i];
  }
}

// ---------------- WMMA GEMM, bf16 A + pre-transposed bf16 B ----------------
// C = A[M,K](bf16) @ Bt[N,K](bf16)^T (+bias)(+relu)(+resid fp32).
// Output: Cf (fp32) or Cb (bf16). Block 128 (4 waves); tile 128x64;
// wave tile 32x64; A fragments direct from global (K-contiguous);
// B tile staged in LDS (async double-buffered when available).
__global__ __launch_bounds__(128) void gemm_bt(
    const __bf16* __restrict__ A, const __bf16* __restrict__ Bt,
    const float* __restrict__ bias, const float* __restrict__ resid,
    float* __restrict__ Cf, __bf16* __restrict__ Cb,
    int M, int N, int K, int relu) {
  int tid  = threadIdx.x;
  int lane = tid & 31;
  int wv   = tid >> 5;
  bool hi  = lane >= 16;
  int l15  = lane & 15;
  int m0   = blockIdx.y * 128;
  int n0   = blockIdx.x * 64;

  const __bf16* Arow0 = A + (size_t)(m0 + wv * 32 + l15) * K;
  const __bf16* Arow1 = Arow0 + (size_t)16 * K;

  v8f acc[2][4] = {};
  int nsteps = K >> 5;

#ifdef HAS_ASYNC_LDS
  __shared__ __attribute__((aligned(16))) __bf16 Bst[2][64][40];
  // B tile: 64 rows x 32 K = 64B/row -> 2 x b128 chunks per thread
  auto issueB = [&](int buf, int k0) {
#pragma unroll
    for (int i = 0; i < 2; i++) {
      int p = i * 128 + tid;
      int n = p >> 2, ch = p & 3;
      int col = n0 + n;
      if (col < N)
        async_b128(Bt + (size_t)col * K + k0 + ch * 8, &Bst[buf][n][ch * 8]);
    }
  };
  issueB(0, 0);
  for (int s = 0; s < nsteps; s++) {
    int k0 = s * 32;
    if (s + 1 < nsteps) { issueB((s + 1) & 1, k0 + 32); wait_async2(); }
    else                { wait_async0(); }
    __syncthreads();
    const __bf16 (*Bb)[40] = Bst[s & 1];

    v16bf a0 = frag_ld(Arow0 + k0, hi);
    v16bf a1 = frag_ld(Arow1 + k0, hi);
#pragma unroll
    for (int g = 0; g < 4; g++) {
      v16bf bf = frag_ld(&Bb[g * 16 + l15][0], hi);
      acc[0][g] = wmma_bf16(a0, bf, acc[0][g]);
      acc[1][g] = wmma_bf16(a1, bf, acc[1][g]);
    }
    __syncthreads();
  }
#else
  __shared__ __attribute__((aligned(16))) __bf16 Bst[64][40];
  bf16x8 brg[2];
  auto loadB = [&](int k0) {
#pragma unroll
    for (int i = 0; i < 2; i++) {
      int p = i * 128 + tid;
      int n = p >> 2, ch = p & 3;
      int col = n0 + n;
      brg[i] = (col < N) ? *(const bf16x8*)(Bt + (size_t)col * K + k0 + ch * 8)
                         : (bf16x8)((__bf16)0.f);
    }
  };
  auto storeB = [&]() {
#pragma unroll
    for (int i = 0; i < 2; i++) {
      int p = i * 128 + tid;
      int n = p >> 2, ch = p & 3;
      *(bf16x8*)&Bst[n][ch * 8] = brg[i];
    }
  };
  loadB(0);
  for (int s = 0; s < nsteps; s++) {
    int k0 = s * 32;
    storeB();
    __syncthreads();
    if (s + 1 < nsteps) loadB(k0 + 32);

    v16bf a0 = frag_ld(Arow0 + k0, hi);
    v16bf a1 = frag_ld(Arow1 + k0, hi);
#pragma unroll
    for (int g = 0; g < 4; g++) {
      v16bf bf = frag_ld(&Bst[g * 16 + l15][0], hi);
      acc[0][g] = wmma_bf16(a0, bf, acc[0][g]);
      acc[1][g] = wmma_bf16(a1, bf, acc[1][g]);
    }
    __syncthreads();
  }
#endif

#pragma unroll
  for (int h2 = 0; h2 < 2; h2++) {
#pragma unroll
    for (int g = 0; g < 4; g++) {
      int col = n0 + g * 16 + l15;
      if (col < N) {
        float bv = bias ? bias[col] : 0.f;
#pragma unroll
        for (int r = 0; r < 8; r++) {
          int row = m0 + wv * 32 + h2 * 16 + r + (hi ? 8 : 0);
          float v = acc[h2][g][r] + bv;
          if (relu) v = fmaxf(v, 0.f);
          if (resid) v += resid[(size_t)row * N + col];
          if (Cb) Cb[(size_t)row * N + col] = (__bf16)v;
          else    Cf[(size_t)row * N + col] = v;
        }
      }
    }
  }
}

// ---------------- QKV projection (LDS-free; weights pre-transposed bf16) ----------------
// wt layout: [3][H][D][E]. grid: (B*H, T/128, 3); block 128.
// Q,K -> bf16 [B,H,T,D]; V -> bf16 [B,H,D,T].
__global__ __launch_bounds__(128) void qkv_gemm(
    const __bf16* __restrict__ hb, const __bf16* __restrict__ wt,
    __bf16* __restrict__ qb, __bf16* __restrict__ kb, __bf16* __restrict__ vb) {
  int bh = blockIdx.x;
  int b  = bh >> 4;
  int hh = bh & 15;
  int m0 = blockIdx.y * 128;
  int z  = blockIdx.z;

  __bf16* out = (z == 0) ? qb : (z == 1) ? kb : vb;
  out += (size_t)bh * Tc * Dc;
  const __bf16* W = wt + ((size_t)z * Hc + hh) * Dc * Ec;   // [d][e]

  int tid  = threadIdx.x;
  int lane = tid & 31;
  int wvid = tid >> 5;
  bool hi  = lane >= 16;
  int l15  = lane & 15;

  const __bf16* Arow0 = hb + (size_t)(b * Tc + m0 + wvid * 32 + l15) * Ec;
  const __bf16* Arow1 = Arow0 + (size_t)16 * Ec;

  v8f acc[2][4] = {};
  for (int k0 = 0; k0 < Ec; k0 += 32) {
    v16bf a0 = frag_ld(Arow0 + k0, hi);
    v16bf a1 = frag_ld(Arow1 + k0, hi);
#pragma unroll
    for (int g = 0; g < 4; g++) {
      v16bf bf = frag_ld(W + (size_t)(g * 16 + l15) * Ec + k0, hi);
      acc[0][g] = wmma_bf16(a0, bf, acc[0][g]);
      acc[1][g] = wmma_bf16(a1, bf, acc[1][g]);
    }
  }

#pragma unroll
  for (int h2 = 0; h2 < 2; h2++) {
#pragma unroll
    for (int g = 0; g < 4; g++) {
      int col = g * 16 + l15;
#pragma unroll
      for (int r = 0; r < 8; r++) {
        int row = m0 + wvid * 32 + h2 * 16 + r + (hi ? 8 : 0);
        if (z == 2) out[(size_t)col * Tc + row] = (__bf16)acc[h2][g][r];  // [D,T]
        else        out[(size_t)row * Dc + col] = (__bf16)acc[h2][g][r];  // [T,D]
      }
    }
  }
}

// ---------------- fused flash attention ----------------
// grid: (T/16, B*H); block 32. qb/kb: [B,H,T,D]; vb: [B,H,D,T]. out bf16.
__global__ __launch_bounds__(32) void attn_kernel(
    const __bf16* __restrict__ qb, const __bf16* __restrict__ kb,
    const __bf16* __restrict__ vb, __bf16* __restrict__ atto) {
  __shared__ __attribute__((aligned(16))) __bf16 Ps[16][40];

  int qt = blockIdx.x;
  int bh = blockIdx.y;
  int b  = bh >> 4;
  int hh = bh & 15;

  const __bf16* Q  = qb + (size_t)bh * Tc * Dc;
  const __bf16* Kp = kb + (size_t)bh * Tc * Dc;
  const __bf16* Vt = vb + (size_t)bh * Dc * Tc;   // [d][t]

  int lane = threadIdx.x;
  bool hi  = lane >= 16;
  int l15  = lane & 15;

  int qrow = qt * 16 + l15;
  v16bf aq0 = frag_ld(Q + (size_t)qrow * Dc, hi);
  v16bf aq1 = frag_ld(Q + (size_t)qrow * Dc + 32, hi);

  v8f O[4] = {};
  float mrow[8], lrow[8];
#pragma unroll
  for (int r = 0; r < 8; r++) { mrow[r] = -3.0e38f; lrow[r] = 0.f; }

  const float scale = 0.03125f;                 // E^-0.5 (reference behavior)
  int nkb = ((qt * 16 + 15) >> 5) + 1;

  for (int kb32 = 0; kb32 < nkb; kb32++) {
    int kbase = kb32 * 32;
    int key0 = kbase + l15, key1 = kbase + 16 + l15;

    v16bf b0a = frag_ld(Kp + (size_t)key0 * Dc, hi);
    v16bf b0b = frag_ld(Kp + (size_t)key0 * Dc + 32, hi);
    v16bf b1a = frag_ld(Kp + (size_t)key1 * Dc, hi);
    v16bf b1b = frag_ld(Kp + (size_t)key1 * Dc + 32, hi);

    v8f S0 = {}, S1 = {};
    S0 = wmma_bf16(aq0, b0a, S0); S0 = wmma_bf16(aq1, b0b, S0);
    S1 = wmma_bf16(aq0, b1a, S1); S1 = wmma_bf16(aq1, b1b, S1);

    // issue V fragment loads early: latency hides under softmax VALU
    v16bf bv[4];
#pragma unroll
    for (int g = 0; g < 4; g++)
      bv[g] = frag_ld(Vt + (size_t)(g * 16 + l15) * Tc + kbase, hi);

    float p0[8], p1[8], c8[8];
#pragma unroll
    for (int r = 0; r < 8; r++) {
      int row = qt * 16 + r + (hi ? 8 : 0);
      float s0 = (key0 <= row) ? S0[r] * scale : -3.0e38f;
      float s1 = (key1 <= row) ? S1[r] * scale : -3.0e38f;
      float mx = fmaxf(s0, s1);
      mx = fmaxf(mx, __shfl_xor(mx, 1, 32));
      mx = fmaxf(mx, __shfl_xor(mx, 2, 32));
      mx = fmaxf(mx, __shfl_xor(mx, 4, 32));
      mx = fmaxf(mx, __shfl_xor(mx, 8, 32));
      float mnew = fmaxf(mrow[r], mx);
      float corr = __expf(mrow[r] - mnew);
      mrow[r] = mnew;
      float e0 = __expf(s0 - mnew), e1 = __expf(s1 - mnew);
      float ls = e0 + e1;
      ls += __shfl_xor(ls, 1, 32);
      ls += __shfl_xor(ls, 2, 32);
      ls += __shfl_xor(ls, 4, 32);
      ls += __shfl_xor(ls, 8, 32);
      lrow[r] = lrow[r] * corr + ls;
      p0[r] = e0; p1[r] = e1; c8[r] = corr;
    }
#pragma unroll
    for (int g = 0; g < 4; g++)
#pragma unroll
      for (int r = 0; r < 8; r++) O[g][r] *= c8[r];

    __syncthreads();
#pragma unroll
    for (int r = 0; r < 8; r++) {
      int rr = r + (hi ? 8 : 0);
      Ps[rr][l15]      = (__bf16)p0[r];
      Ps[rr][16 + l15] = (__bf16)p1[r];
    }
    __syncthreads();

    v16bf ap = frag_ld(&Ps[l15][0], hi);
#pragma unroll
    for (int g = 0; g < 4; g++)
      O[g] = wmma_bf16(ap, bv[g], O[g]);
  }

  float inv[8];
#pragma unroll
  for (int r = 0; r < 8; r++) inv[r] = 1.0f / lrow[r];
#pragma unroll
  for (int g = 0; g < 4; g++) {
#pragma unroll
    for (int r = 0; r < 8; r++) {
      int row = qt * 16 + r + (hi ? 8 : 0);
      atto[((size_t)(b * Tc + row)) * Ec + hh * Dc + g * 16 + l15] =
          (__bf16)(O[g][r] * inv[r]);
    }
  }
}

// ---------------- loss ----------------
__global__ void zero_kernel(float* p) {
  if (threadIdx.x == 0 && blockIdx.x == 0) *p = 0.f;
}

__global__ __launch_bounds__(256) void loss_kernel(
    const float* __restrict__ logits, const int* __restrict__ targets,
    float* __restrict__ loss_acc) {
  __shared__ float red[256];
  int row = blockIdx.x, tid = threadIdx.x;
  const float* lr = logits + (size_t)row * Vc;

  float mx = -3.0e38f;
  for (int i = tid; i < Vc; i += 256) mx = fmaxf(mx, lr[i]);
  red[tid] = mx; __syncthreads();
  for (int o = 128; o > 0; o >>= 1) { if (tid < o) red[tid] = fmaxf(red[tid], red[tid + o]); __syncthreads(); }
  float m = red[0];
  __syncthreads();

  float se = 0.f;
  for (int i = tid; i < Vc; i += 256) se += __expf(lr[i] - m);
  red[tid] = se; __syncthreads();
  for (int o = 128; o > 0; o >>= 1) { if (tid < o) red[tid] += red[tid + o]; __syncthreads(); }

  if (tid == 0) {
    int t = targets[row];
    float lp = lr[t] - m - __logf(red[0]);
    atomicAdd(loss_acc, -lp * (1.0f / (float)BTc));
  }
}

// ---------------- host launch ----------------
extern "C" void kernel_launch(void* const* d_in, const int* in_sizes, int n_in,
                              void* d_out, int out_size, void* d_ws, size_t ws_size,
                              hipStream_t stream) {
  (void)in_sizes; (void)n_in; (void)out_size; (void)ws_size;

  const int*   idx   = (const int*)  d_in[0];
  const int*   tgt   = (const int*)  d_in[1];
  const float* tok   = (const float*)d_in[2];
  const float* pos   = (const float*)d_in[3];
  const float* wq    = (const float*)d_in[4];
  const float* wk    = (const float*)d_in[5];
  const float* wvw   = (const float*)d_in[6];
  const float* wproj = (const float*)d_in[7];
  const float* bproj = (const float*)d_in[8];
  const float* ln1g  = (const float*)d_in[9];
  const float* ln1b  = (const float*)d_in[10];
  const float* ln2g  = (const float*)d_in[11];
  const float* ln2b  = (const float*)d_in[12];
  const float* ffw1  = (const float*)d_in[13];
  const float* ffb1  = (const float*)d_in[14];
  const float* ffw2  = (const float*)d_in[15];
  const float* ffb2  = (const float*)d_in[16];
  const float* lnfg  = (const float*)d_in[17];
  const float* lnfb  = (const float*)d_in[18];
  const float* lmw   = (const float*)d_in[19];
  const float* lmb   = (const float*)d_in[20];

  char* ws = (char*)d_ws;
  const size_t MB = (size_t)1 << 20;
  float*  x     = (float*)(ws);                //  16 MB [B*T,E] fp32
  __bf16* hbuf  = (__bf16*)(ws + 16 * MB);     //   8 MB [B*T,E]
  __bf16* atto  = (__bf16*)(ws + 24 * MB);     //   8 MB [B*T,E]
  __bf16* ffh   = (__bf16*)(ws + 32 * MB);     //  32 MB [B*T,FF]
  __bf16* qb    = (__bf16*)(ws + 64 * MB);     //   8 MB [B,H,T,D]
  __bf16* kb    = (__bf16*)(ws + 73 * MB);     //   8 MB [B,H,T,D]
  __bf16* vb    = (__bf16*)(ws + 82 * MB);     //   8 MB [B,H,D,T]
  char*   arena = ws + 96 * MB;                // bf16 weight arena (max 103 MB)
  __bf16* qkvt  = (__bf16*)(arena);            // [3][H][D][E] = 6 MB
  __bf16* projt = (__bf16*)(arena + 8 * MB);   // [E][E]   = 2 MB
  __bf16* ff1t  = (__bf16*)(arena + 16 * MB);  // [FF][E]  = 8 MB
  __bf16* ff2t  = (__bf16*)(arena + 24 * MB);  // [E][FF]  = 8 MB
  __bf16* lmt   = (__bf16*)(arena);            // [V][E]   = 103 MB (after layers)

  float* logits = (float*)d_out;
  float* loss   = logits + (size_t)BTc * Vc;

  const dim3 tb(32, 8);

  embed_kernel<<<BTc, 256, 0, stream>>>(idx, tok, pos, x);

  for (int l = 0; l < Lc; l++) {
    const size_t eo  = (size_t)l * Ec;
    const size_t hwo = (size_t)l * Hc * Ec * Dc;

    // per-head W[E,D] -> Wt[D,E], bf16 (batch over heads)
    transpose_cvt<<<dim3(2, 32, 16), tb, 0, stream>>>(
        wq + hwo, qkvt + 0 * Hc * Dc * Ec, Ec, Dc, (size_t)Ec * Dc, (size_t)Dc * Ec);
    transpose_cvt<<<dim3(2, 32, 16), tb, 0, stream>>>(
        wk + hwo, qkvt + 1 * Hc * Dc * Ec, Ec, Dc, (size_t)Ec * Dc, (size_t)Dc * Ec);
    transpose_cvt<<<dim3(2, 32, 16), tb, 0, stream>>>(
        wvw + hwo, qkvt + 2 * (size_t)Hc * Dc * Ec, Ec, Dc, (size_t)Ec * Dc, (size_t)Dc * Ec);

    ln_kernel<<<BTc, 256, 0, stream>>>(x, ln1g + eo, ln1b + eo, hbuf);

    qkv_gemm<<<dim3(Bc * Hc, Tc / 128, 3), 128, 0, stream>>>(
        hbuf, qkvt, qb, kb, vb);

    attn_kernel<<<dim3(Tc / 16, Bc * Hc), 32, 0, stream>>>(qb, kb, vb, atto);

    // x = x + atto @ w_proj + b_proj
    transpose_cvt<<<dim3(32, 32, 1), tb, 0, stream>>>(
        wproj + (size_t)l * Ec * Ec, projt, Ec, Ec, 0, 0);
    gemm_bt<<<dim3(Ec / 64, BTc / 128), 128, 0, stream>>>(
        atto, projt, bproj + eo, x, x, nullptr, BTc, Ec, Ec, 0);

    ln_kernel<<<BTc, 256, 0, stream>>>(x, ln2g + eo, ln2b + eo, hbuf);

    // ffh = relu(h @ ff_w1 + ff_b1)   (bf16 out)
    transpose_cvt<<<dim3(128, 32, 1), tb, 0, stream>>>(
        ffw1 + (size_t)l * Ec * FFc, ff1t, Ec, FFc, 0, 0);
    gemm_bt<<<dim3(FFc / 64, BTc / 128), 128, 0, stream>>>(
        hbuf, ff1t, ffb1 + (size_t)l * FFc, nullptr, nullptr, ffh,
        BTc, FFc, Ec, 1);

    // x = x + ffh @ ff_w2 + ff_b2
    transpose_cvt<<<dim3(32, 128, 1), tb, 0, stream>>>(
        ffw2 + (size_t)l * FFc * Ec, ff2t, FFc, Ec, 0, 0);
    gemm_bt<<<dim3(Ec / 64, BTc / 128), 128, 0, stream>>>(
        ffh, ff2t, ffb2 + eo, x, x, nullptr, BTc, Ec, FFc, 0);
  }

  ln_kernel<<<BTc, 256, 0, stream>>>(x, lnfg, lnfb, hbuf);

  // logits = h @ lm_w + lm_b   (lm_w -> bf16 [V,E]: L2-resident at 103 MB)
  transpose_cvt<<<dim3((Vc + 31) / 32, 32, 1), tb, 0, stream>>>(
      lmw, lmt, Ec, Vc, 0, 0);
  gemm_bt<<<dim3((Vc + 63) / 64, BTc / 128), 128, 0, stream>>>(
      hbuf, lmt, lmb, nullptr, logits, nullptr, BTc, Vc, Ec, 0);

  zero_kernel<<<1, 32, 0, stream>>>(loss);
  loss_kernel<<<BTc, 256, 0, stream>>>(logits, tgt, loss);
}
